// LocalFeatureAggregation_23519240913434
// MI455X (gfx1250) — compile-verified, matrix operand order checked by reading, not run on hardware
//
#include <hip/hip_runtime.h>

// ---------------- types ----------------
typedef __attribute__((ext_vector_type(16))) __bf16 v16bf;
typedef __attribute__((ext_vector_type(8)))  float  v8f;
typedef __attribute__((ext_vector_type(4)))  unsigned int u32x4; // trivial 16B vector (union-safe)

union Frag {
    unsigned int u[8];
    u32x4        q[2];
    v16bf        v;
};

// ---------------- constants ----------------
#define EPSV 1e-5f
#define NPTS 8192
#define BN_TOTAL 32768       // B*N
#define CAT_LD 264           // padded row (in bf16 elems) for 256-ch cat tile

// workspace layout (bytes)
#define W_ATTN_OFF 0u              // 16nt*8kt*32lane*16h bf16 = 128KB
#define W_O_OFF    131072u         // 128KB
#define W_S_OFF    262144u         // 16nt*4kt*32*16 bf16 = 64KB
#define W_N_OFF    327680u         // 10*128 f32 folded = 5KB
#define T_N_OFF    332800u         // 128 f32
#define T_O_OFF    333312u         // 256 f32
#define T_S_OFF    334336u         // 256 f32
#define FEAT_OFF   335360u         // 4M bf16 = 8MB
#define POOL_OFF   8723968u        // 32768*256 bf16 = 16MB  (total ~24.3MB)

// ---------------- helpers ----------------
__device__ __forceinline__ unsigned short f2bf(float f) {
    unsigned int u = __float_as_uint(f);
    u += 0x7FFFu + ((u >> 16) & 1u);        // round-to-nearest-even
    return (unsigned short)(u >> 16);
}
__device__ __forceinline__ float bf2f(unsigned short h) {
    return __uint_as_float(((unsigned int)h) << 16);
}
// K-offset inside a 16-bit 16x32 fragment: VGPR v (0..7), lane-half hi (0/1)
__device__ __forceinline__ int kofs(int v, int hi) {
    return 16 * (v >> 2) + 8 * hi + 2 * (v & 3);
}

// ============================================================
// Kernel 0a: fold BN into weights, convert+swizzle to bf16 fragments
// ============================================================
__global__ __launch_bounds__(256) void prep_weights(
    const float* __restrict__ w_n,  const float* __restrict__ b_n,
    const float* __restrict__ g_n,  const float* __restrict__ beta_n,
    const float* __restrict__ rm_n, const float* __restrict__ rv_n,
    const float* __restrict__ w_attn,
    const float* __restrict__ w_o,  const float* __restrict__ b_o,
    const float* __restrict__ g_o,  const float* __restrict__ beta_o,
    const float* __restrict__ rm_o, const float* __restrict__ rv_o,
    const float* __restrict__ w_s,  const float* __restrict__ b_s,
    const float* __restrict__ g_s,  const float* __restrict__ beta_s,
    const float* __restrict__ rm_s, const float* __restrict__ rv_s,
    char* __restrict__ ws)
{
    int gid = blockIdx.x * blockDim.x + threadIdx.x;
    unsigned short* wattn_sw = (unsigned short*)(ws + W_ATTN_OFF);
    unsigned short* wo_sw    = (unsigned short*)(ws + W_O_OFF);
    unsigned short* ws_sw    = (unsigned short*)(ws + W_S_OFF);
    float* wn_f = (float*)(ws + W_N_OFF);
    float* tn   = (float*)(ws + T_N_OFF);
    float* to   = (float*)(ws + T_O_OFF);
    float* tsf  = (float*)(ws + T_S_OFF);

    if (gid < 65536) {                          // w_attn -> swizzled bf16 (no BN)
        int j = gid;
        int h = j & 15, lane = (j >> 4) & 31, kt = (j >> 9) & 7, nt = j >> 12;
        int v = h >> 1, pr = h & 1;
        int K = kt * 32 + kofs(v, lane >> 4) + pr;
        int c = nt * 16 + (lane & 15);
        wattn_sw[j] = f2bf(w_attn[K * 256 + c]);
    } else if (gid < 131072) {                  // w_o * s_o -> swizzled bf16
        int j = gid - 65536;
        int h = j & 15, lane = (j >> 4) & 31, kt = (j >> 9) & 7, nt = j >> 12;
        int v = h >> 1, pr = h & 1;
        int K = kt * 32 + kofs(v, lane >> 4) + pr;
        int c = nt * 16 + (lane & 15);
        float s = g_o[c] * rsqrtf(rv_o[c] + EPSV);
        wo_sw[j] = f2bf(w_o[K * 256 + c] * s);
    } else if (gid < 163840) {                  // w_s * s_s -> swizzled bf16 (K=128 -> 4 kt)
        int j = gid - 131072;
        int h = j & 15, lane = (j >> 4) & 31, kt = (j >> 9) & 3, nt = j >> 11;
        int v = h >> 1, pr = h & 1;
        int K = kt * 32 + kofs(v, lane >> 4) + pr;
        int c = nt * 16 + (lane & 15);
        float s = g_s[c] * rsqrtf(rv_s[c] + EPSV);
        ws_sw[j] = f2bf(w_s[K * 256 + c] * s);
    } else if (gid < 165120) {                  // w_n * s_n (fp32, VALU path)
        int j = gid - 163840;
        int c = j & 127;
        float s = g_n[c] * rsqrtf(rv_n[c] + EPSV);
        wn_f[j] = w_n[j] * s;
    } else if (gid < 165248) {                  // t_n
        int c = gid - 165120;
        float s = g_n[c] * rsqrtf(rv_n[c] + EPSV);
        tn[c] = (b_n[c] - rm_n[c]) * s + beta_n[c];
    } else if (gid < 165504) {                  // t_o
        int c = gid - 165248;
        float s = g_o[c] * rsqrtf(rv_o[c] + EPSV);
        to[c] = (b_o[c] - rm_o[c]) * s + beta_o[c];
    } else if (gid < 165760) {                  // t_s
        int c = gid - 165504;
        float s = g_s[c] * rsqrtf(rv_s[c] + EPSV);
        tsf[c] = (b_s[c] - rm_s[c]) * s + beta_s[c];
    }
}

// ============================================================
// Kernel 0b: feature fp32 -> bf16 (row-major copy)
// ============================================================
__global__ __launch_bounds__(256) void cvt_feature(
    const float4* __restrict__ f, uint2* __restrict__ out)
{
    int gid = blockIdx.x * blockDim.x + threadIdx.x;
    if (gid >= 1048576) return;     // 4*8192*128 / 4
    float4 v = f[gid];
    uint2 o;
    o.x = (unsigned)f2bf(v.x) | ((unsigned)f2bf(v.y) << 16);
    o.y = (unsigned)f2bf(v.z) | ((unsigned)f2bf(v.w) << 16);
    out[gid] = o;
}

// ============================================================
// Kernel 1: gather + neighbor MLP -> cat tiles in LDS,
//           attn GEMM (WMMA bf16) + softmax over K + pooling.
// 16 points per workgroup; each wave owns 2 points so every
// w_attn B-fragment load feeds two WMMAs (2x B reuse).
// ============================================================
__global__ __launch_bounds__(256) void lfa_attn_pool(
    const unsigned short* __restrict__ featbf,
    const float* __restrict__ rawnb,
    const int* __restrict__ nidx,
    const unsigned short* __restrict__ wattn_sw,
    const float* __restrict__ wn_f,
    const float* __restrict__ tn,
    unsigned short* __restrict__ pooled)
{
    __shared__ unsigned short s_cat[16 * 16 * CAT_LD]; // 132KB
    __shared__ float s_wn[1280];
    __shared__ float s_tn[128];

    const int tid = threadIdx.x;
    const int blk = blockIdx.x;

    for (int i = tid; i < 1280; i += 256) s_wn[i] = wn_f[i];
    if (tid < 128) s_tn[tid] = tn[tid];
    __syncthreads();

    // -------- phase 1: build cat[16][16][256] (bf16), 1 thread per row --------
    {
        int p = tid >> 4, k = tid & 15;          // 256 rows
        int gp = blk * 16 + p;
        int b = gp >> 13, n = gp & 8191;
        int idx = nidx[(((b << 13) | n) << 4) + k];

        // gather: copy 128 bf16 channels (256B) from feature row
        const u32x4* src = (const u32x4*)(featbf + ((size_t)((b << 13) + idx)) * 128);
        u32x4* dst = (u32x4*)(s_cat + (p * 16 + k) * CAT_LD);
        #pragma unroll
        for (int i = 0; i < 16; i++) dst[i] = src[i];

        // neighbor MLP: raw[10] @ wn_f[10,128] + t_n, leaky_relu
        const float* rp = rawnb + ((size_t)((((b << 13) | n) << 4) + k)) * 10;
        float rv[10];
        #pragma unroll
        for (int g = 0; g < 10; g++) rv[g] = rp[g];
        for (int c = 0; c < 128; c++) {
            float acc = s_tn[c];
            #pragma unroll
            for (int g = 0; g < 10; g++) acc = fmaf(rv[g], s_wn[g * 128 + c], acc);
            acc = (acc >= 0.f) ? acc : 0.2f * acc;
            s_cat[(p * 16 + k) * CAT_LD + 128 + c] = f2bf(acc);
        }
    }
    __syncthreads();

    // -------- phase 2: per-wave attn GEMM (2 points) + softmax + pool --------
    const int wave = tid >> 5, lane = tid & 31;
    const int p0 = wave * 2, p1 = wave * 2 + 1;
    const int gp0 = blk * 16 + p0, gp1 = blk * 16 + p1;
    const int M = lane & 15, hi = lane >> 4;
    const int ch_l = lane & 15;

    // preload A-fragments for both points (neighbor rows, K = 256 channels)
    Frag a0[8], a1[8];
    #pragma unroll
    for (int kt = 0; kt < 8; kt++) {
        #pragma unroll
        for (int v = 0; v < 8; v++) {
            int K = kt * 32 + kofs(v, hi);
            a0[kt].u[v] = *(const unsigned int*)&s_cat[(p0 * 16 + M) * CAT_LD + K];
            a1[kt].u[v] = *(const unsigned int*)&s_cat[(p1 * 16 + M) * CAT_LD + K];
        }
    }

    const u32x4* wB = (const u32x4*)wattn_sw;

    for (int nt = 0; nt < 16; nt++) {
        v8f acc0 = {0.f, 0.f, 0.f, 0.f, 0.f, 0.f, 0.f, 0.f};
        v8f acc1 = {0.f, 0.f, 0.f, 0.f, 0.f, 0.f, 0.f, 0.f};
        #pragma unroll
        for (int kt = 0; kt < 8; kt++) {
            Frag bf;
            int bi = ((nt * 8 + kt) * 32 + lane) * 2;
            bf.q[0] = wB[bi];
            bf.q[1] = wB[bi + 1];
            acc0 = __builtin_amdgcn_wmma_f32_16x16x32_bf16(
                false, a0[kt].v, false, bf.v, (short)0, acc0, false, false);
            acc1 = __builtin_amdgcn_wmma_f32_16x16x32_bf16(
                false, a1[kt].v, false, bf.v, (short)0, acc1, false, false);
        }
        int ch = nt * 16 + ch_l;

        // ---- point 0: softmax over 16 neighbors + weighted pool ----
        {
            float mx = acc0[0];
            #pragma unroll
            for (int r = 1; r < 8; r++) mx = fmaxf(mx, acc0[r]);
            mx = fmaxf(mx, __shfl_xor(mx, 16, 32));
            float e[8]; float sm = 0.f;
            #pragma unroll
            for (int r = 0; r < 8; r++) { e[r] = __expf(acc0[r] - mx); sm += e[r]; }
            sm += __shfl_xor(sm, 16, 32);
            float inv = 1.f / sm;
            float part = 0.f;
            #pragma unroll
            for (int r = 0; r < 8; r++) {
                float cv = bf2f(s_cat[(p0 * 16 + (hi * 8 + r)) * CAT_LD + ch]);
                part = fmaf(e[r], cv, part);
            }
            part *= inv;
            part += __shfl_xor(part, 16, 32);
            if (hi == 0) pooled[(size_t)gp0 * 256 + ch] = f2bf(part);
        }
        // ---- point 1 ----
        {
            float mx = acc1[0];
            #pragma unroll
            for (int r = 1; r < 8; r++) mx = fmaxf(mx, acc1[r]);
            mx = fmaxf(mx, __shfl_xor(mx, 16, 32));
            float e[8]; float sm = 0.f;
            #pragma unroll
            for (int r = 0; r < 8; r++) { e[r] = __expf(acc1[r] - mx); sm += e[r]; }
            sm += __shfl_xor(sm, 16, 32);
            float inv = 1.f / sm;
            float part = 0.f;
            #pragma unroll
            for (int r = 0; r < 8; r++) {
                float cv = bf2f(s_cat[(p1 * 16 + (hi * 8 + r)) * CAT_LD + ch]);
                part = fmaf(e[r], cv, part);
            }
            part *= inv;
            part += __shfl_xor(part, 16, 32);
            if (hi == 0) pooled[(size_t)gp1 * 256 + ch] = f2bf(part);
        }
    }
}

// ============================================================
// Kernel 2: out = lrelu( feat@w_s' + pooled@w_o' + (t_s+t_o) )
// 16 points (one M-tile) per workgroup; wave w does ntiles 2w,2w+1
// ============================================================
__global__ __launch_bounds__(256) void lfa_out(
    const unsigned short* __restrict__ featbf,
    const unsigned short* __restrict__ pooled,
    const unsigned short* __restrict__ ws_sw,
    const unsigned short* __restrict__ wo_sw,
    const float* __restrict__ tsf,
    const float* __restrict__ to,
    float* __restrict__ out)
{
    const int tid = threadIdx.x, blk = blockIdx.x;
    const int wave = tid >> 5, lane = tid & 31;
    const int M = lane & 15, hi = lane >> 4;
    const int pt = blk * 16 + M;

    Frag as_[4], ao[8];
    #pragma unroll
    for (int kt = 0; kt < 4; kt++) {
        #pragma unroll
        for (int v = 0; v < 8; v++) {
            int K = kt * 32 + kofs(v, hi);
            as_[kt].u[v] = *(const unsigned int*)&featbf[(size_t)pt * 128 + K];
        }
    }
    #pragma unroll
    for (int kt = 0; kt < 8; kt++) {
        #pragma unroll
        for (int v = 0; v < 8; v++) {
            int K = kt * 32 + kofs(v, hi);
            ao[kt].u[v] = *(const unsigned int*)&pooled[(size_t)pt * 256 + K];
        }
    }

    const u32x4* wS = (const u32x4*)ws_sw;
    const u32x4* wO = (const u32x4*)wo_sw;

    #pragma unroll
    for (int t = 0; t < 2; t++) {
        int nt = wave * 2 + t;
        v8f acc = {0.f, 0.f, 0.f, 0.f, 0.f, 0.f, 0.f, 0.f};
        #pragma unroll
        for (int kt = 0; kt < 4; kt++) {
            Frag bf; int bi = ((nt * 4 + kt) * 32 + lane) * 2;
            bf.q[0] = wS[bi]; bf.q[1] = wS[bi + 1];
            acc = __builtin_amdgcn_wmma_f32_16x16x32_bf16(
                false, as_[kt].v, false, bf.v, (short)0, acc, false, false);
        }
        #pragma unroll
        for (int kt = 0; kt < 8; kt++) {
            Frag bf; int bi = ((nt * 8 + kt) * 32 + lane) * 2;
            bf.q[0] = wO[bi]; bf.q[1] = wO[bi + 1];
            acc = __builtin_amdgcn_wmma_f32_16x16x32_bf16(
                false, ao[kt].v, false, bf.v, (short)0, acc, false, false);
        }
        int ch = nt * 16 + (lane & 15);
        float tb = tsf[ch] + to[ch];
        #pragma unroll
        for (int r = 0; r < 8; r++) {
            float v = acc[r] + tb;
            v = (v >= 0.f) ? v : 0.2f * v;
            out[(size_t)(blk * 16 + hi * 8 + r) * 256 + ch] = v;
        }
    }
}

// ============================================================
extern "C" void kernel_launch(void* const* d_in, const int* in_sizes, int n_in,
                              void* d_out, int out_size, void* d_ws, size_t ws_size,
                              hipStream_t stream)
{
    (void)in_sizes; (void)n_in; (void)out_size; (void)ws_size;

    const float* feature = (const float*)d_in[0];
    const float* rawnb   = (const float*)d_in[1];
    const int*   nidx    = (const int*)d_in[2];
    const float* w_n     = (const float*)d_in[3];
    const float* b_n     = (const float*)d_in[4];
    const float* g_n     = (const float*)d_in[5];
    const float* beta_n  = (const float*)d_in[6];
    const float* rm_n    = (const float*)d_in[7];
    const float* rv_n    = (const float*)d_in[8];
    const float* w_attn  = (const float*)d_in[9];
    const float* w_o     = (const float*)d_in[10];
    const float* b_o     = (const float*)d_in[11];
    const float* g_o     = (const float*)d_in[12];
    const float* beta_o  = (const float*)d_in[13];
    const float* rm_o    = (const float*)d_in[14];
    const float* rv_o    = (const float*)d_in[15];
    const float* w_s     = (const float*)d_in[16];
    const float* b_s     = (const float*)d_in[17];
    const float* g_s     = (const float*)d_in[18];
    const float* beta_s  = (const float*)d_in[19];
    const float* rm_s    = (const float*)d_in[20];
    const float* rv_s    = (const float*)d_in[21];

    char*  ws  = (char*)d_ws;
    float* out = (float*)d_out;

    unsigned short* featbf = (unsigned short*)(ws + FEAT_OFF);
    unsigned short* pooled = (unsigned short*)(ws + POOL_OFF);

    prep_weights<<<648, 256, 0, stream>>>(
        w_n, b_n, g_n, beta_n, rm_n, rv_n,
        w_attn,
        w_o, b_o, g_o, beta_o, rm_o, rv_o,
        w_s, b_s, g_s, beta_s, rm_s, rv_s, ws);

    cvt_feature<<<4096, 256, 0, stream>>>((const float4*)feature, (uint2*)featbf);

    lfa_attn_pool<<<BN_TOTAL / 16, 256, 0, stream>>>(
        featbf, rawnb, nidx,
        (const unsigned short*)(ws + W_ATTN_OFF),
        (const float*)(ws + W_N_OFF),
        (const float*)(ws + T_N_OFF),
        pooled);

    lfa_out<<<BN_TOTAL / 16, 256, 0, stream>>>(
        featbf, pooled,
        (const unsigned short*)(ws + W_S_OFF),
        (const unsigned short*)(ws + W_O_OFF),
        (const float*)(ws + T_S_OFF),
        (const float*)(ws + T_O_OFF),
        out);
}
